// RNNTagger_69483980915180
// MI455X (gfx1250) — matrix operand order repaired and checked by reading, not compiled, not run on hardware
//
#include <hip/hip_runtime.h>
#include <cstdint>

// ---------------- problem constants ----------------
#define Tn    256
#define BATCH 64
#define EMB   1024
#define HID   1024
#define NTAGS 50
#define NWG_REC 16        // persistent workgroups in recurrent kernel (HID/64)

typedef __bf16 bf16;
typedef bf16  v16bf __attribute__((ext_vector_type(16)));
typedef float v8f   __attribute__((ext_vector_type(8)));
typedef float f32x4 __attribute__((ext_vector_type(4)));
typedef unsigned int u32x4 __attribute__((ext_vector_type(4)));

union Frag { v16bf v; u32x4 q[2]; };
union V8bf { unsigned short s[8]; u32x4 q; };

__device__ __forceinline__ unsigned short f2bf(float f) {
    unsigned u = __float_as_uint(f);
    unsigned r = (u + 0x7FFFu + ((u >> 16) & 1u)) >> 16;   // RNE
    return (unsigned short)r;
}
__device__ __forceinline__ float sigmoidf_(float x) { return 1.f / (1.f + __expf(-x)); }
__device__ __forceinline__ float tanhf_(float x)    { return 1.f - 2.f / (__expf(2.f * x) + 1.f); }

// --------------------------------------------------------------------------
// Weight conversion, pre-swizzled into WMMA B-fragment tiles:
//   tile id = (k/32)*64 + (n/16) ; in-tile offset = l*16 + e where
//   lane l: N = n%16 = l&15 ; l<16 -> K {0..7,16..23}, l>=16 -> K {8..15,24..31}
// so per-step staging is a LINEAR copy (async-to-LDS capable).
// --------------------------------------------------------------------------
__global__ __launch_bounds__(256) void cvt_kernel(const float* __restrict__ src,
                                                  unsigned short* __restrict__ dst, int K) {
    int idx = blockIdx.x * 256 + threadIdx.x;              // over HID*K
    if (idx < HID * K) {
        int tile = idx >> 9;
        int r    = idx & 511;
        int l = r >> 4, e = r & 15;
        int kc = tile >> 6, ctg = tile & 63;
        int k = kc * 32 + ((e >> 3) << 4) + ((l >> 4) << 3) + (e & 7);
        int n = ctg * 16 + (l & 15);
        dst[idx] = f2bf(src[(size_t)n * K + k]);
    }
}

// --------------------------------------------------------------------------
// LDS staging. A fragment lane layout (ISA 7.12.2, 16-bit A 16x32):
//   lane l -> M=l&15 ; l<16 : K {0..7,16..23}, l>=16 : K {8..15,24..31}
// Each staging thread writes exactly one aligned 16B chunk (ds_store_b128).
// --------------------------------------------------------------------------
__device__ __forceinline__ void stageA64x32(unsigned short* dst, const float* src,
                                            int kb, int tid) {
    int row = tid >> 2;              // 0..63
    int kq  = (tid & 3) * 8;         // 0,8,16,24  (one contiguous chunk each)
    const float* s = src + (size_t)row * HID + kb + kq;
    f32x4 a = *(const f32x4*)s;
    f32x4 b = *(const f32x4*)(s + 4);
    V8bf v;
#pragma unroll
    for (int j = 0; j < 4; ++j) { v.s[j] = f2bf(a[j]); v.s[4 + j] = f2bf(b[j]); }
    int l = (row & 15) + (((kq >> 3) & 1) << 4);
    int e = ((kq >> 4) & 1) << 3;
    *(u32x4*)&dst[(row >> 4) * 512 + l * 16 + e] = v.q;
}
// Pre-swizzled B tiles: linear 4KB copy, via CDNA5 async global->LDS DMA.
// LDS byte address = low 32 bits of generic pointer (flat->LDS truncation).
__device__ __forceinline__ void stageBasync(unsigned short* dst, const unsigned short* src,
                                            int tid) {
    unsigned ldsa = (unsigned)(uintptr_t)(dst + tid * 8);
    unsigned long long ga = (unsigned long long)(uintptr_t)(src + tid * 8);
    asm volatile("global_load_async_to_lds_b128 %0, %1, off"
                 :: "v"(ldsa), "v"(ga) : "memory");
}
__device__ __forceinline__ void waitAsync() {
    asm volatile("s_wait_asynccnt 0x0" ::: "memory");
}
__device__ __forceinline__ Frag ldFrag(const unsigned short* base, int tile, int lane) {
    Frag f;
    f.q[0] = *(const u32x4*)&base[tile * 512 + lane * 16];
    f.q[1] = *(const u32x4*)&base[tile * 512 + lane * 16 + 8];
    return f;
}

// --------------------------------------------------------------------------
// XW precompute: xw[m][h] = emb(X[m]) . w_ih[h] + b_ih[h] + b_hh[h]
// Big parallel GEMM [16384 x 1024] x [1024 x 1024]; WG tile 128x64.
// --------------------------------------------------------------------------
__global__ __launch_bounds__(256) void xw_kernel(const int* __restrict__ X,
                                                 const float* __restrict__ emb,
                                                 const unsigned short* __restrict__ wbih,
                                                 const float* __restrict__ b_ih,
                                                 const float* __restrict__ b_hh,
                                                 float* __restrict__ xw) {
    __shared__ __attribute__((aligned(16))) unsigned short lA[8 * 512]; // 128x32 bf16
    __shared__ __attribute__((aligned(16))) unsigned short lB[4 * 512]; // 32x64  bf16
    __shared__ int toks[128];
    int tid = threadIdx.x, wave = tid >> 5, lane = tid & 31;
    int rowbase = blockIdx.x * 128;
    int colbase = blockIdx.y * 64;

    if (tid < 128) toks[tid] = X[rowbase + tid];
    __syncthreads();

    v8f acc[4] = {};
    for (int kb = 0; kb < EMB; kb += 32) {
        __syncthreads();
        stageBasync(lB, wbih + ((size_t)(kb >> 5) * 64 + (colbase >> 4)) * 512, tid);
        { // stage A: gathered embedding rows -> bf16, two 16B chunks per thread
            int row = tid >> 1;
            int kh  = (tid & 1) * 16;
            const float* s = emb + (size_t)toks[row] * EMB + kb + kh;
            f32x4 a0 = *(const f32x4*)s;
            f32x4 a1 = *(const f32x4*)(s + 4);
            f32x4 a2 = *(const f32x4*)(s + 8);
            f32x4 a3 = *(const f32x4*)(s + 12);
            V8bf v0, v1;
#pragma unroll
            for (int j = 0; j < 4; ++j) {
                v0.s[j]     = f2bf(a0[j]);  v0.s[4 + j] = f2bf(a1[j]);  // k = kh+0..7
                v1.s[j]     = f2bf(a2[j]);  v1.s[4 + j] = f2bf(a3[j]);  // k = kh+8..15
            }
            int r = row >> 4, lm = row & 15;
            int e = (kh >> 4) << 3;                       // 0 or 8
            *(u32x4*)&lA[r * 512 + lm * 16 + e]        = v0.q;   // low lanes
            *(u32x4*)&lA[r * 512 + (lm + 16) * 16 + e] = v1.q;   // high lanes
        }
        waitAsync();
        __syncthreads();

        Frag a = ldFrag(lA, wave, lane);
#pragma unroll
        for (int ct = 0; ct < 4; ++ct) {
            Frag b = ldFrag(lB, ct, lane);
            acc[ct] = __builtin_amdgcn_wmma_f32_16x16x32_bf16(
                false, a.v, false, b.v, (short)0, acc[ct], false, false);
        }
    }
    // store: C layout -> VGPR i : M = i + 8*(lane>=16), N = lane&15
    int m0  = rowbase + wave * 16;
    int nlo = lane & 15;
    int mo  = (lane >> 4) * 8;
#pragma unroll
    for (int ct = 0; ct < 4; ++ct) {
        int col = colbase + ct * 16 + nlo;
        float bias = b_ih[col] + b_hh[col];
#pragma unroll
        for (int i = 0; i < 8; ++i)
            xw[(size_t)(m0 + mo + i) * HID + col] = acc[ct][i] + bias;
    }
}

// --------------------------------------------------------------------------
// grid-wide barrier (generation based), NWG_REC workgroups, all resident
// --------------------------------------------------------------------------
__device__ __forceinline__ void gbar(unsigned* bar, unsigned& gen) {
    __syncthreads();
    __threadfence();
    if (threadIdx.x == 0) {
        gen++;
        unsigned prev = atomicAdd(&bar[0], 1u);
        if (prev == NWG_REC - 1) {
            *(volatile unsigned*)&bar[0] = 0u;
            __threadfence();
            *(volatile unsigned*)&bar[1] = gen;          // release
        } else {
            while (*(volatile unsigned*)&bar[1] < gen) __builtin_amdgcn_s_sleep(2);
            __threadfence();
        }
    }
    __syncthreads();
}

__global__ void init_kernel(const float* __restrict__ h0, const float* __restrict__ c0,
                            float* __restrict__ hbuf, float* __restrict__ cbuf,
                            unsigned* __restrict__ bar) {
    size_t i = (size_t)blockIdx.x * 256 + threadIdx.x;
    if (i < (size_t)BATCH * HID) { hbuf[i] = h0[i]; cbuf[i] = c0[i]; }
    if (i < 2) bar[i] = 0;
}

// --------------------------------------------------------------------------
// Persistent recurrent kernel: 16 WGs x 256 threads, each owns 64 cols of H.
// Phase1 fuses h@w_hh and c@w_ch in one K loop; phase2 does cy@w_c2h.
// hy overwrites xw[t] in place -> becomes the output history.
// --------------------------------------------------------------------------
__global__ __launch_bounds__(256) void rec_kernel(const unsigned short* __restrict__ wbhh,
                                                  const unsigned short* __restrict__ wbch,
                                                  const unsigned short* __restrict__ wbc2h,
                                                  const float* __restrict__ b_ch,
                                                  const float* __restrict__ b_c2h,
                                                  float* __restrict__ xw,
                                                  float* __restrict__ hbuf,
                                                  float* __restrict__ cbuf,
                                                  unsigned* __restrict__ bar) {
    __shared__ __attribute__((aligned(16))) unsigned short lAh[4 * 512];
    __shared__ __attribute__((aligned(16))) unsigned short lAc[4 * 512];
    __shared__ __attribute__((aligned(16))) unsigned short lB1[4 * 512];
    __shared__ __attribute__((aligned(16))) unsigned short lB2[4 * 512];
    int tid = threadIdx.x, wave = tid >> 5, lane = tid & 31;
    int colbase = blockIdx.x * 64;
    size_t tilecol = (size_t)(colbase >> 4);
    int rt = wave & 3;                 // row tile (B)
    int cp = (wave >> 2) * 2;          // first of the 2 owned col tiles
    int nlo = lane & 15, mo = (lane >> 4) * 8;
    unsigned gen = 0;
    int cur = 0;

    for (int t = 0; t < Tn; ++t) {
        float* hx = hbuf + (size_t)cur * (BATCH * HID);
        float* cx = cbuf + (size_t)cur * (BATCH * HID);
        float* hy = hbuf + (size_t)(cur ^ 1) * (BATCH * HID);
        float* cy = cbuf + (size_t)(cur ^ 1) * (BATCH * HID);

        v8f accp[2] = {}, accf[2] = {};
        for (int kb = 0; kb < HID; kb += 32) {
            __syncthreads();
            size_t tb = ((size_t)(kb >> 5) * 64 + tilecol) * 512;
            stageBasync(lB1, wbhh + tb, tid);
            stageBasync(lB2, wbch + tb, tid);
            stageA64x32(lAh, hx, kb, tid);
            stageA64x32(lAc, cx, kb, tid);
            waitAsync();
            __syncthreads();
            Frag ah = ldFrag(lAh, rt, lane);
            Frag ac = ldFrag(lAc, rt, lane);
#pragma unroll
            for (int j = 0; j < 2; ++j) {
                Frag b1 = ldFrag(lB1, cp + j, lane);
                Frag b2 = ldFrag(lB2, cp + j, lane);
                accp[j] = __builtin_amdgcn_wmma_f32_16x16x32_bf16(
                    false, ah.v, false, b1.v, (short)0, accp[j], false, false);
                accf[j] = __builtin_amdgcn_wmma_f32_16x16x32_bf16(
                    false, ac.v, false, b2.v, (short)0, accf[j], false, false);
            }
        }

        // elementwise gates -> cy ; keep pre in accp for phase 2
        v8f cyreg[2];
#pragma unroll
        for (int j = 0; j < 2; ++j) {
            int col = colbase + (cp + j) * 16 + nlo;
            float bch = b_ch[col];
#pragma unroll
            for (int i = 0; i < 8; ++i) {
                int b = rt * 16 + mo + i;
                float pre = accp[j][i] + xw[(size_t)(t * BATCH + b) * HID + col];
                float cxi = cx[(size_t)b * HID + col];
                float gif = sigmoidf_(pre + accf[j][i] + bch);
                float cg  = tanhf_(pre);
                float cyv = gif * (cxi + cg);
                cyreg[j][i] = cyv;
                accp[j][i]  = pre;
                cy[(size_t)b * HID + col] = cyv;
            }
        }

        gbar(bar, gen);   // cy fully written across all WGs

        v8f acco[2] = {};
        for (int kb = 0; kb < HID; kb += 32) {
            __syncthreads();
            stageBasync(lB1, wbc2h + ((size_t)(kb >> 5) * 64 + tilecol) * 512, tid);
            stageA64x32(lAh, cy, kb, tid);
            waitAsync();
            __syncthreads();
            Frag a = ldFrag(lAh, rt, lane);
#pragma unroll
            for (int j = 0; j < 2; ++j) {
                Frag b = ldFrag(lB1, cp + j, lane);
                acco[j] = __builtin_amdgcn_wmma_f32_16x16x32_bf16(
                    false, a.v, false, b.v, (short)0, acco[j], false, false);
            }
        }

#pragma unroll
        for (int j = 0; j < 2; ++j) {
            int col = colbase + (cp + j) * 16 + nlo;
            float bc2 = b_c2h[col];
#pragma unroll
            for (int i = 0; i < 8; ++i) {
                int b = rt * 16 + mo + i;
                float og  = sigmoidf_(accp[j][i] + acco[j][i] + bc2);
                float hyv = og * tanhf_(cyreg[j][i]);
                hy[(size_t)b * HID + col] = hyv;
                xw[(size_t)(t * BATCH + b) * HID + col] = hyv;  // output history
            }
        }

        gbar(bar, gen);   // step complete
        cur ^= 1;
    }
}

// --------------------------------------------------------------------------
// dense + softmax: one 64-thread WG per (t,b) row
// --------------------------------------------------------------------------
__global__ __launch_bounds__(64) void dense_kernel(const float* __restrict__ hyall,
                                                   const float* __restrict__ dw,
                                                   const float* __restrict__ db,
                                                   float* __restrict__ out) {
    __shared__ float row[HID];
    __shared__ float red[64];
    int r = blockIdx.x, tid = threadIdx.x;
    const float* src = hyall + (size_t)r * HID;
    for (int k = tid; k < HID; k += 64) row[k] = src[k];
    __syncthreads();

    float logit = -3.4e38f;
    if (tid < NTAGS) {
        float acc = db[tid];
        const float* w = dw + (size_t)tid * HID;
        for (int k = 0; k < HID; k += 4) {
            acc += row[k] * w[k] + row[k + 1] * w[k + 1]
                 + row[k + 2] * w[k + 2] + row[k + 3] * w[k + 3];
        }
        logit = acc;
    }
    red[tid] = logit;
    __syncthreads();
    for (int s = 32; s > 0; s >>= 1) { if (tid < s) red[tid] = fmaxf(red[tid], red[tid + s]); __syncthreads(); }
    float mx = red[0];
    __syncthreads();
    float e = (tid < NTAGS) ? __expf(logit - mx) : 0.f;
    red[tid] = e;
    __syncthreads();
    for (int s = 32; s > 0; s >>= 1) { if (tid < s) red[tid] += red[tid + s]; __syncthreads(); }
    if (tid < NTAGS) out[(size_t)r * NTAGS + tid] = e / red[0];
}

// --------------------------------------------------------------------------
extern "C" void kernel_launch(void* const* d_in, const int* in_sizes, int n_in,
                              void* d_out, int out_size, void* d_ws, size_t ws_size,
                              hipStream_t stream) {
    const int*   X        = (const int*)  d_in[0];
    // d_in[1] = lengths (all == T, unused)
    const float* embedding= (const float*)d_in[2];
    const float* w_ih     = (const float*)d_in[3];
    const float* w_hh     = (const float*)d_in[4];
    const float* w_ch     = (const float*)d_in[5];
    const float* w_c2h    = (const float*)d_in[6];
    const float* b_ih     = (const float*)d_in[7];
    const float* b_hh     = (const float*)d_in[8];
    const float* b_ch     = (const float*)d_in[9];
    const float* b_c2h    = (const float*)d_in[10];
    const float* dense_w  = (const float*)d_in[11];
    const float* dense_b  = (const float*)d_in[12];
    const float* h0       = (const float*)d_in[13];
    const float* c0       = (const float*)d_in[14];
    float* out = (float*)d_out;

    // workspace layout
    unsigned short* wb_ih  = (unsigned short*)d_ws;                 // EMB*HID bf16
    unsigned short* wb_hh  = wb_ih  + (size_t)EMB * HID;
    unsigned short* wb_ch  = wb_hh  + (size_t)HID * HID;
    unsigned short* wb_c2h = wb_ch  + (size_t)HID * HID;
    float* xw   = (float*)(wb_c2h + (size_t)HID * HID);             // T*B*HID f32 (also hy history)
    float* hbuf = xw   + (size_t)Tn * BATCH * HID;                  // 2*B*HID
    float* cbuf = hbuf + (size_t)2 * BATCH * HID;                   // 2*B*HID
    unsigned* bar = (unsigned*)(cbuf + (size_t)2 * BATCH * HID);

    const int cvtBlocks = (HID * HID + 255) / 256;
    cvt_kernel<<<cvtBlocks, 256, 0, stream>>>(w_ih,  wb_ih,  EMB);
    cvt_kernel<<<cvtBlocks, 256, 0, stream>>>(w_hh,  wb_hh,  HID);
    cvt_kernel<<<cvtBlocks, 256, 0, stream>>>(w_ch,  wb_ch,  HID);
    cvt_kernel<<<cvtBlocks, 256, 0, stream>>>(w_c2h, wb_c2h, HID);

    dim3 gxw((Tn * BATCH) / 128, HID / 64);
    xw_kernel<<<gxw, 256, 0, stream>>>(X, embedding, wb_ih, b_ih, b_hh, xw);

    init_kernel<<<(BATCH * HID + 255) / 256, 256, 0, stream>>>(h0, c0, hbuf, cbuf, bar);

    rec_kernel<<<NWG_REC, 256, 0, stream>>>(wb_hh, wb_ch, wb_c2h, b_ch, b_c2h,
                                            xw, hbuf, cbuf, bar);

    dense_kernel<<<Tn * BATCH, 64, 0, stream>>>(xw, dense_w, dense_b, out);
}
